// DeepGraphConv_Surv_3624952398639
// MI455X (gfx1250) — compile-verified, matrix operand order checked by reading, not run on hardware
//
#include <hip/hip_runtime.h>
#include <hip/hip_bf16.h>
#include <math.h>

#define NN     30000
#define EE     480000
#define FIN    1024
#define HH     256
#define NCLS   4
#define NNODE  8

typedef __attribute__((ext_vector_type(16))) __bf16 v16bf;
typedef __attribute__((ext_vector_type(8)))  __bf16 v8bf;
typedef __attribute__((ext_vector_type(8)))  float  v8f;

#define ACT_NONE 0
#define ACT_RELU 1
#define ACT_TANH 2
#define ACT_SIG  3

// ---------------- elementwise helpers ----------------

__global__ void k_copy_f4(const float* __restrict__ s, float* __restrict__ d, size_t n4) {
    size_t i = (size_t)blockIdx.x * blockDim.x + threadIdx.x;
    if (i < n4) ((float4*)d)[i] = ((const float4*)s)[i];
}

__global__ void k_f32_to_bf16(const float* __restrict__ s, __bf16* __restrict__ d, size_t n) {
    size_t i = (size_t)blockIdx.x * blockDim.x + threadIdx.x;
    if (i < n) d[i] = (__bf16)s[i];
}

__global__ void k_zero_f32(float* __restrict__ d, int n) {
    int i = blockIdx.x * blockDim.x + threadIdx.x;
    if (i < n) d[i] = 0.0f;
}

// ---------------- weight pack: fp32 [K,Nc] -> bf16 WMMA-B-fragment order ----------------
// packed[((c*nt + t)*32 + lane)*16 + e] = (bf16)W[k,n]
__global__ void k_pack_w(const float* __restrict__ W, __bf16* __restrict__ pw, int K, int Nc) {
    size_t idx = (size_t)blockIdx.x * blockDim.x + threadIdx.x;
    if (idx >= (size_t)K * Nc) return;
    int e    = (int)(idx & 15);
    int l    = (int)((idx >> 4) & 31);
    int nt   = Nc >> 4;
    size_t g = idx >> 9;                 // c*nt + t
    int t    = (int)(g % nt);
    int c    = (int)(g / nt);
    int lrow = l & 15, half = l >> 4;
    int v    = e >> 1, w = e & 1;
    int kk   = (v < 4) ? (2 * v + w) : (16 + 2 * (v - 4) + w);
    int k    = c * 32 + (half ? 8 : 0) + kk;
    int n    = t * 16 + lrow;
    pw[idx]  = (__bf16)W[(size_t)k * Nc + n];
}

// ---------------- graph aggregation: hsum[dst] += x[src] ----------------
__global__ void k_edge_scatter(const float* __restrict__ x,
                               const long long* __restrict__ src,
                               const long long* __restrict__ dst,
                               float* __restrict__ hsum, int F) {
    int c = threadIdx.x;               // float4 index
    int e = blockIdx.x;
    if (c >= (F >> 2)) return;
    long long s = src[e];
    long long d = dst[e];
    const float4 v = *(const float4*)(x + (size_t)s * F + 4 * c);
    float* o = hsum + (size_t)d * F + 4 * c;
    atomicAdd(o + 0, v.x);
    atomicAdd(o + 1, v.y);
    atomicAdd(o + 2, v.z);
    atomicAdd(o + 3, v.w);
}

// ---------------- WMMA bf16 GEMM: C = act(A[M,K] @ W[K,Nc] + bias) ----------------
// A: row-major bf16, staged per-block through LDS with async copies (double-buffered).
// PW: weights pre-packed in B-fragment order. 256 threads = 8 waves; wave w computes
// C tiles (blockIdx.x, w*2) and (blockIdx.x, w*2+1), sharing one A fragment.
// REQUIRES: M % 16 == 0, Nc == 16*16, grid covers exactly (no partial tiles).
__global__ void k_gemm_bf16_wmma(const __bf16* __restrict__ A, const __bf16* __restrict__ PW,
                                 const float* __restrict__ bias, float* __restrict__ C,
                                 int M, int K, int Nc, int act) {
    // double-buffered A chunk: 16 rows x 32 K-values bf16 = 1KB per buffer
    __shared__ __align__(16) __bf16 sA[2][512];

    const int tid   = threadIdx.x;
    const int lane  = tid & 31;
    const int wave  = tid >> 5;
    const int half  = lane >> 4;
    const int lrow  = lane & 15;
    const int tileM = blockIdx.x;
    const int tN0   = wave * 2;               // first of two N-tiles (Nc == 256)
    const int nt    = Nc >> 4;
    const int nchunks = K >> 5;

    // async-stage addressing: each of 256 threads copies 4B of the 1KB chunk
    const int o    = tid * 4;                 // byte offset in chunk tile
    const int srow = o >> 6;                  // source row 0..15
    const int scol = o & 63;                  // byte within row
    const char* gbase = (const char*)(A + (size_t)(tileM * 16 + srow) * K) + scol;
    const unsigned lds0 = (unsigned)(unsigned long long)(const void*)&sA[0][0];
    const unsigned lds1 = (unsigned)(unsigned long long)(const void*)&sA[1][0];

    #define STAGE_A(cc, buf)                                                     \
        do {                                                                     \
            unsigned long long ga = (unsigned long long)(gbase + (size_t)(cc) * 64); \
            unsigned la = ((buf) ? lds1 : lds0) + (unsigned)o;                   \
            asm volatile("global_load_async_to_lds_b32 %0, %1, off"              \
                         :: "v"(la), "v"(ga) : "memory");                        \
        } while (0)

    STAGE_A(0, 0);

    v8f acc0 = {}, acc1 = {};
    for (int c = 0; c < nchunks; ++c) {
        const int buf = c & 1;
        asm volatile("s_wait_asynccnt 0x0" ::: "memory");
        __syncthreads();                      // chunk c resident; prior reads of buf done
        if (c + 1 < nchunks) STAGE_A(c + 1, buf ^ 1);

        // A fragment from LDS: row lrow, two 16B runs (half selects K sub-range)
        const char* sb = (const char*)&sA[buf][0] + lrow * 64 + half * 16;
        v8bf alo = *(const v8bf*)(sb);
        v8bf ahi = *(const v8bf*)(sb + 32);
        v16bf af = __builtin_shufflevector(alo, ahi,
                     0, 1, 2, 3, 4, 5, 6, 7, 8, 9, 10, 11, 12, 13, 14, 15);

        // B fragments: contiguous 32B per lane from the packed layout (global/L2)
        const __bf16* pb = PW + (((size_t)c * nt + tN0) * 32 + lane) * 16;
        v16bf b0 = *(const v16bf*)pb;
        v16bf b1 = *(const v16bf*)(pb + 512);          // next N-tile
        __builtin_prefetch(pb + (size_t)nt * 512, 0, 0);  // next k-chunk of B

        acc0 = __builtin_amdgcn_wmma_f32_16x16x32_bf16(
                   false, af, false, b0, (short)0, acc0, false, false);
        acc1 = __builtin_amdgcn_wmma_f32_16x16x32_bf16(
                   false, af, false, b1, (short)0, acc1, false, false);

        __syncthreads();                      // protect double buffer for iter c+2 staging
    }
    #undef STAGE_A

    // C/D layout: VGPR r, lane l -> row r + half*8, col lrow
    #pragma unroll
    for (int s = 0; s < 2; ++s) {
        const int cn  = (tN0 + s) * 16 + lrow;
        const float bv = bias ? bias[cn] : 0.0f;
        const v8f& a  = s ? acc1 : acc0;
        #pragma unroll
        for (int r = 0; r < 8; ++r) {
            int cm = tileM * 16 + r + half * 8;
            float v = a[r] + bv;
            if (act == ACT_RELU)      v = fmaxf(v, 0.0f);
            else if (act == ACT_TANH) v = tanhf(v);
            else if (act == ACT_SIG)  v = 1.0f / (1.0f + __expf(-v));
            C[(size_t)cm * Nc + cn] = v;
        }
    }
}

// ---------------- per-node head: Y_node = x3 @ nodeW + nodeb ----------------
__global__ void k_node_head(const float* __restrict__ x3, const float* __restrict__ W,
                            const float* __restrict__ b, float* __restrict__ out) {
    int idx = blockIdx.x * blockDim.x + threadIdx.x;
    if (idx >= NN * NNODE) return;
    int n = idx >> 3, c = idx & 7;
    const float* row = x3 + (size_t)n * HH;
    float s = b[c];
    #pragma unroll 4
    for (int j = 0; j < HH; ++j) s += row[j] * W[j * NNODE + c];
    out[idx] = s;
}

// ---------------- attention score: A[n] = sum_j a*b*cW[j] + cb ----------------
__global__ void k_attn(const float* __restrict__ a, const float* __restrict__ b,
                       const float* __restrict__ cW, const float* __restrict__ cb,
                       float* __restrict__ att, float* __restrict__ a_raw) {
    int i = blockIdx.x * blockDim.x + threadIdx.x;
    if (i >= NN) return;
    const float* ar = a + (size_t)i * HH;
    const float* br = b + (size_t)i * HH;
    float s = cb[0];
    #pragma unroll 4
    for (int j = 0; j < HH; ++j) s += ar[j] * br[j] * cW[j];
    att[i]   = s;
    a_raw[i] = s;
}

// ---------------- single-block softmax stats: scal[0]=max, scal[1]=sum(exp) ----------------
__global__ void k_softmax_stats(const float* __restrict__ A, float* __restrict__ scal, int n) {
    __shared__ float sm[1024];
    int t = threadIdx.x;
    float m = -INFINITY;
    for (int i = t; i < n; i += 1024) m = fmaxf(m, A[i]);
    sm[t] = m; __syncthreads();
    for (int s = 512; s > 0; s >>= 1) { if (t < s) sm[t] = fmaxf(sm[t], sm[t + s]); __syncthreads(); }
    float mx = sm[0]; __syncthreads();
    float acc = 0.0f;
    for (int i = t; i < n; i += 1024) acc += __expf(A[i] - mx);
    sm[t] = acc; __syncthreads();
    for (int s = 512; s > 0; s >>= 1) { if (t < s) sm[t] += sm[t + s]; __syncthreads(); }
    if (t == 0) { scal[0] = mx; scal[1] = sm[0]; }
}

__global__ void k_softmax_w(const float* __restrict__ A, const float* __restrict__ scal,
                            float* __restrict__ w, int n) {
    int i = blockIdx.x * blockDim.x + threadIdx.x;
    if (i < n) w[i] = __expf(A[i] - scal[0]) / scal[1];
}

// ---------------- attention pooling: pool[j] += sum_chunk w[n]*x3[n,j] ----------------
__global__ void k_pool(const float* __restrict__ w, const float* __restrict__ x3,
                       float* __restrict__ pool, int n) {
    int j  = threadIdx.x;
    int r0 = blockIdx.x * 128;
    int r1 = min(r0 + 128, n);
    float s = 0.0f;
    for (int r = r0; r < r1; ++r) s += w[r] * x3[(size_t)r * HH + j];
    atomicAdd(&pool[j], s);
}

// ---------------- rho: hv = relu(pool @ rW + rb), one block of 256 ----------------
__global__ void k_rho(const float* __restrict__ pool, const float* __restrict__ rW,
                      const float* __restrict__ rb, float* __restrict__ hv) {
    __shared__ float sp[HH];
    int j = threadIdx.x;
    sp[j] = pool[j];
    __syncthreads();
    float s = rb[j];
    #pragma unroll 4
    for (int i = 0; i < HH; ++i) s += sp[i] * rW[i * HH + j];
    hv[j] = fmaxf(s, 0.0f);
}

// ---------------- final head: logits, softmax, argmax, hazards->cumprod ----------------
__global__ void k_head(const float* __restrict__ hv, const float* __restrict__ clsW,
                       const float* __restrict__ clsb, float* __restrict__ out) {
    if (threadIdx.x != 0 || blockIdx.x != 0) return;
    float lg[NCLS];
    for (int c = 0; c < NCLS; ++c) {
        float s = clsb[c];
        for (int j = 0; j < HH; ++j) s += hv[j] * clsW[j * NCLS + c];
        lg[c] = s;
        out[c] = s;                               // logits
    }
    float mx = lg[0];
    for (int c = 1; c < NCLS; ++c) mx = fmaxf(mx, lg[c]);
    float e[NCLS], se = 0.0f;
    for (int c = 0; c < NCLS; ++c) { e[c] = __expf(lg[c] - mx); se += e[c]; }
    for (int c = 0; c < NCLS; ++c) out[4 + c] = e[c] / se;   // Y_prob
    int am = 0;
    for (int c = 1; c < NCLS; ++c) if (lg[c] > lg[am]) am = c;
    out[8] = (float)am;                                       // Y_hat
    float S = 1.0f;
    for (int c = 0; c < NCLS; ++c) {
        float hz = 1.0f / (1.0f + __expf(-lg[c]));
        S *= (1.0f - hz);
        out[9 + NN + c] = S;                                  // S
    }
}

// ---------------- orchestration ----------------

extern "C" void kernel_launch(void* const* d_in, const int* in_sizes, int n_in,
                              void* d_out, int out_size, void* d_ws, size_t ws_size,
                              hipStream_t stream) {
    const float*     x    = (const float*)d_in[0];
    const long long* ei   = (const long long*)d_in[1];
    const long long* esrc = ei;
    const long long* edst = ei + EE;
    const float* c1W1 = (const float*)d_in[2];  const float* c1b1 = (const float*)d_in[3];
    const float* c1W2 = (const float*)d_in[4];  const float* c1b2 = (const float*)d_in[5];
    const float* c2W1 = (const float*)d_in[6];  const float* c2b1 = (const float*)d_in[7];
    const float* c2W2 = (const float*)d_in[8];  const float* c2b2 = (const float*)d_in[9];
    const float* c3W1 = (const float*)d_in[10]; const float* c3b1 = (const float*)d_in[11];
    const float* c3W2 = (const float*)d_in[12]; const float* c3b2 = (const float*)d_in[13];
    const float* aW   = (const float*)d_in[14]; const float* ab   = (const float*)d_in[15];
    const float* bW   = (const float*)d_in[16]; const float* bb   = (const float*)d_in[17];
    const float* cW   = (const float*)d_in[18]; const float* cb   = (const float*)d_in[19];
    const float* rW   = (const float*)d_in[20]; const float* rb   = (const float*)d_in[21];
    const float* clsW = (const float*)d_in[22]; const float* clsb = (const float*)d_in[23];
    const float* ndW  = (const float*)d_in[24]; const float* ndb  = (const float*)d_in[25];

    const size_t NF = (size_t)NN * FIN;
    const size_t NH = (size_t)NN * HH;
    char* p = (char*)d_ws;
    float*  hsum = (float*)p;  p += NF * sizeof(float);
    __bf16* xb   = (__bf16*)p; p += NF * sizeof(__bf16);
    float*  hbuf = (float*)p;  p += NH * sizeof(float);
    float*  xbuf = (float*)p;  p += NH * sizeof(float);
    float*  abuf = (float*)p;  p += NH * sizeof(float);
    float*  bbuf = (float*)p;  p += NH * sizeof(float);
    __bf16* wb   = (__bf16*)p; p += (size_t)FIN * HH * sizeof(__bf16);
    float*  att  = (float*)p;  p += NN * sizeof(float);
    float*  swt  = (float*)p;  p += NN * sizeof(float);
    float*  pool = (float*)p;  p += HH * sizeof(float);
    float*  hv   = (float*)p;  p += HH * sizeof(float);
    float*  scal = (float*)p;  p += 8 * sizeof(float);

    float* out       = (float*)d_out;
    float* out_araw  = out + 9;
    float* out_ynode = out + 9 + NN + NCLS;   // offset 30013

    const int TPB = 256;
    const dim3 ggrid(NN / 16, 1);             // exact: 1875 M-tiles, 8 waves x 2 N-tiles

    auto cvt = [&](const float* s, __bf16* d, size_t n) {
        k_f32_to_bf16<<<(unsigned)((n + TPB - 1) / TPB), TPB, 0, stream>>>(s, d, n);
    };
    auto packw = [&](const float* W, int K) {
        size_t n = (size_t)K * HH;
        k_pack_w<<<(unsigned)((n + TPB - 1) / TPB), TPB, 0, stream>>>(W, wb, K, HH);
    };

    // ---------- GIN layer 1 (F_IN=1024 -> H) ----------
    k_copy_f4<<<(unsigned)((NF / 4 + TPB - 1) / TPB), TPB, 0, stream>>>(x, hsum, NF / 4);
    k_edge_scatter<<<EE, FIN / 4, 0, stream>>>(x, esrc, edst, hsum, FIN);
    cvt(hsum, xb, NF);
    packw(c1W1, FIN);
    k_gemm_bf16_wmma<<<ggrid, TPB, 0, stream>>>(xb, wb, c1b1, hbuf, NN, FIN, HH, ACT_RELU);
    cvt(hbuf, xb, NH);
    packw(c1W2, HH);
    k_gemm_bf16_wmma<<<ggrid, TPB, 0, stream>>>(xb, wb, c1b2, xbuf, NN, HH, HH, ACT_RELU);

    // ---------- GIN layer 2 ----------
    k_copy_f4<<<(unsigned)((NH / 4 + TPB - 1) / TPB), TPB, 0, stream>>>(xbuf, hsum, NH / 4);
    k_edge_scatter<<<EE, HH / 4, 0, stream>>>(xbuf, esrc, edst, hsum, HH);
    cvt(hsum, xb, NH);
    packw(c2W1, HH);
    k_gemm_bf16_wmma<<<ggrid, TPB, 0, stream>>>(xb, wb, c2b1, hbuf, NN, HH, HH, ACT_RELU);
    cvt(hbuf, xb, NH);
    packw(c2W2, HH);
    k_gemm_bf16_wmma<<<ggrid, TPB, 0, stream>>>(xb, wb, c2b2, xbuf, NN, HH, HH, ACT_RELU);

    // ---------- GIN layer 3 ----------
    k_copy_f4<<<(unsigned)((NH / 4 + TPB - 1) / TPB), TPB, 0, stream>>>(xbuf, hsum, NH / 4);
    k_edge_scatter<<<EE, HH / 4, 0, stream>>>(xbuf, esrc, edst, hsum, HH);
    cvt(hsum, xb, NH);
    packw(c3W1, HH);
    k_gemm_bf16_wmma<<<ggrid, TPB, 0, stream>>>(xb, wb, c3b1, hbuf, NN, HH, HH, ACT_RELU);
    cvt(hbuf, xb, NH);
    packw(c3W2, HH);
    k_gemm_bf16_wmma<<<ggrid, TPB, 0, stream>>>(xb, wb, c3b2, xbuf, NN, HH, HH, ACT_RELU);
    // xbuf == x3 == h_path

    // ---------- node head ----------
    k_node_head<<<(NN * NNODE + TPB - 1) / TPB, TPB, 0, stream>>>(xbuf, ndW, ndb, out_ynode);

    // ---------- gated attention ----------
    cvt(xbuf, xb, NH);
    packw(aW, HH);
    k_gemm_bf16_wmma<<<ggrid, TPB, 0, stream>>>(xb, wb, ab, abuf, NN, HH, HH, ACT_TANH);
    packw(bW, HH);
    k_gemm_bf16_wmma<<<ggrid, TPB, 0, stream>>>(xb, wb, bb, bbuf, NN, HH, HH, ACT_SIG);
    k_attn<<<(NN + TPB - 1) / TPB, TPB, 0, stream>>>(abuf, bbuf, cW, cb, att, out_araw);

    // ---------- softmax over nodes + pooling ----------
    k_softmax_stats<<<1, 1024, 0, stream>>>(att, scal, NN);
    k_softmax_w<<<(NN + TPB - 1) / TPB, TPB, 0, stream>>>(att, scal, swt, NN);
    k_zero_f32<<<1, HH, 0, stream>>>(pool, HH);
    k_pool<<<(NN + 127) / 128, HH, 0, stream>>>(swt, xbuf, pool, NN);

    // ---------- rho + classifier head ----------
    k_rho<<<1, HH, 0, stream>>>(pool, rW, rb, hv);
    k_head<<<1, 1, 0, stream>>>(hv, clsW, clsb, out);
}
